// GraphMessageLayer_33423435497891
// MI455X (gfx1250) — compile-verified, matrix operand order checked by reading, not run on hardware
//
#include <hip/hip_runtime.h>
#include <hip/hip_bf16.h>

typedef __bf16  v16bf __attribute__((ext_vector_type(16)));
typedef float   v8f   __attribute__((ext_vector_type(8)));
typedef float   v2f   __attribute__((ext_vector_type(2)));

#define WAVES_PER_WG 4
#define STRIDE 264            // ushort elements per LDS row (256 + 8 pad -> 528B, 16B aligned, rows 4 banks apart)
#define WAVE_LDS_BYTES (8448 + 8448 + 512)   // bufA + bufB + geo

__device__ __forceinline__ unsigned short f2bf(float f) {
    unsigned u = __builtin_bit_cast(unsigned, f);
    u = (u + 0x7FFFu + ((u >> 16) & 1u)) >> 16;
    return (unsigned short)u;
}
// Pack two f32 -> two bf16 (RNE) in one dword; uses v_cvt_pk_bf16_f32 when available.
__device__ __forceinline__ unsigned pack_bf16(float a, float b) {
#if __has_builtin(__builtin_amdgcn_cvt_pk_bf16_f32)
    auto pk = __builtin_amdgcn_cvt_pk_bf16_f32(a, b);
    return __builtin_bit_cast(unsigned, pk);
#else
    return (unsigned)f2bf(a) | ((unsigned)f2bf(b) << 16);
#endif
}
__device__ __forceinline__ float bf2f_lo(unsigned u) {   // low bf16 of a packed dword
    return __builtin_bit_cast(float, u << 16);
}
__device__ __forceinline__ float bf2f_hi(unsigned u) {   // high bf16 of a packed dword
    return __builtin_bit_cast(float, u & 0xFFFF0000u);
}

struct Pack32 { uint4 lo; uint4 hi; };

// A fragment (16x32 bf16): lane L holds row m=L%16; lanes 0-15: K {0..7,16..23}, lanes 16-31: K {8..15,24..31}
__device__ __forceinline__ v16bf load_a_frag(const unsigned short* buf, int lm, int half, int kt) {
    const unsigned short* p = buf + lm * STRIDE + kt * 32 + half * 8;
    Pack32 u;
    u.lo = *(const uint4*)(p);        // K base..base+7
    u.hi = *(const uint4*)(p + 16);   // K base+16..base+23
    return __builtin_bit_cast(v16bf, u);
}

// B fragment pre-swizzled in workspace: 512 ushorts per (kt,nt) tile, lane-contiguous 32B
__device__ __forceinline__ v16bf load_b_frag(const unsigned short* frag, int tile, int lane) {
    const unsigned short* q = frag + tile * 512 + lane * 16;
    Pack32 u;
    u.lo = *(const uint4*)(q);
    u.hi = *(const uint4*)(q + 8);
    return __builtin_bit_cast(v16bf, u);
}

// Store a D-tile row pair (rows v, v+1 of the same column) from one packed dword.
__device__ __forceinline__ void store_pair(unsigned short* p0, unsigned short* p1, float a, float b) {
    const unsigned u = pack_bf16(a, b);
    *p0 = (unsigned short)u;          // ds_store_b16
    *p1 = (unsigned short)(u >> 16);  // ds_store_b16_d16_hi
}

// Convert f32 row-major W[K][Ncols] into bf16 WMMA-B fragments (mirror of the ISA A layout).
__global__ void swizzle_bf16_frag(const float* __restrict__ W, unsigned short* __restrict__ o,
                                  int K, int Ncols, int total) {
    int t = blockIdx.x * blockDim.x + threadIdx.x;
    if (t >= total) return;
    int lane = t & 31;
    int tile = t >> 5;
    int ntiles = Ncols >> 4;
    int nt = tile % ntiles;
    int kt = tile / ntiles;
    int half = lane >> 4;
    int n = nt * 16 + (lane & 15);
    unsigned* dst = (unsigned*)(o + tile * 512 + lane * 16);
#pragma unroll
    for (int v = 0; v < 8; ++v) {
        int base = kt * 32 + ((v < 4) ? (half * 8 + v * 2) : (16 + half * 8 + (v - 4) * 2));
        dst[v] = pack_bf16(W[(base + 0) * Ncols + n], W[(base + 1) * Ncols + n]);
    }
}

__global__ __launch_bounds__(128) void fused_graph_msg(
    const float* __restrict__ x, const int* __restrict__ nbr, const float* __restrict__ boxes,
    const float* __restrict__ We1, const float* __restrict__ be1,
    const float* __restrict__ be2,
    const float* __restrict__ bm1, const float* __restrict__ bm2,
    const float* __restrict__ Wa, const float* __restrict__ ba,
    const unsigned short* __restrict__ fWe2,
    const unsigned short* __restrict__ fWm1,
    const unsigned short* __restrict__ fWm2,
    float* __restrict__ out, int n_nodes)
{
    extern __shared__ char smem_raw[];
    const int tid  = threadIdx.x;
    const int wave = tid >> 5;
    const int lane = tid & 31;
    const int half = lane >> 4;
    const int lm   = lane & 15;

    char* wbase = smem_raw + wave * WAVE_LDS_BYTES;
    unsigned short* bufA = (unsigned short*)wbase;            // msg_in: [16][STRIDE] bf16 (cols 0..255 used)
    unsigned short* bufB = (unsigned short*)(wbase + 8448);   // h, then t
    float*          geo  = (float*)(wbase + 16896);           // [16][8]: dist,scale,cd,sd,alpha,jbits

    const int i = blockIdx.x * WAVES_PER_WG + wave;
    const bool active = (i < n_nodes);

    // ---- Phase 1: per-edge geometry (lanes 0..15 = the 16 neighbors) ----
    if (active && lane < 16) {
        const int row = lane;
        const int j = nbr[i * 16 + row];
        const float bi0 = boxes[i * 5 + 0], bi1 = boxes[i * 5 + 1];
        const float bi2 = boxes[i * 5 + 2], bi3 = boxes[i * 5 + 3], bi4 = boxes[i * 5 + 4];
        const float bj0 = boxes[j * 5 + 0], bj1 = boxes[j * 5 + 1];
        const float bj2 = boxes[j * 5 + 2], bj3 = boxes[j * 5 + 3], bj4 = boxes[j * 5 + 4];
        const float dth = bi4 - bj4;
        const float cd = cosf(dth), sd = sinf(dth);
        const float dx = bi0 - bj0, dy = bi1 - bj1;
        const float dist = sqrtf(dx * dx + dy * dy);
        const float mi = fminf(bi2, bi3), mj = fminf(bj2, bj3);
        const float scale = logf(fmaxf(mi / mj, 1e-6f));
        float* g = geo + row * 8;
        g[0] = dist; g[1] = scale; g[2] = cd; g[3] = sd;
        ((int*)g)[5] = j;
    }
    __syncthreads();

    // ---- Phase 2: gather + rotate neighbor features -> bufA[:,0:128] (bf16) ----
    if (active) {
        for (int row = 0; row < 16; ++row) {
            const float* g = geo + row * 8;
            const int j = ((const int*)g)[5];
            const float cd = g[2], sd = g[3];
            const float4 v = *(const float4*)(x + (size_t)j * 128 + lane * 4);
            float f0 = v.x, f1 = v.y;
            if (lane == 0) {                       // cols 0,1 carry the rotated pair
                const float r0 = cd * f0 - sd * f1;
                const float r1 = sd * f0 + cd * f1;
                f0 = r0; f1 = r1;
            }
            uint2 pk;
            pk.x = pack_bf16(f0, f1);
            pk.y = pack_bf16(v.z, v.w);
            *(uint2*)(bufA + row * STRIDE + lane * 4) = pk;
        }
    }
    __syncthreads();

    // ---- Phase 3: edge MLP layer 1:  h = relu(e @ We1 + be1)  (16x4 @ 4x128), f32 WMMA ----
    if (active) {
        const int k0 = half ? 2 : 0;
#if __has_builtin(__builtin_amdgcn_wmma_f32_16x16x4_f32)
        v2f a; a.x = geo[lm * 8 + k0]; a.y = geo[lm * 8 + k0 + 1];
        for (int nt = 0; nt < 8; ++nt) {
            const int n = nt * 16 + lm;
            v2f b; b.x = We1[(k0 + 0) * 128 + n]; b.y = We1[(k0 + 1) * 128 + n];
            v8f c = {};
            c = __builtin_amdgcn_wmma_f32_16x16x4_f32(false, a, false, b, (short)0, c, false, false);
            const float bias = be1[n];
#pragma unroll
            for (int v = 0; v < 8; v += 2)
                store_pair(bufB + (v + 0 + 8 * half) * STRIDE + n,
                           bufB + (v + 1 + 8 * half) * STRIDE + n,
                           fmaxf(c[v] + bias, 0.f), fmaxf(c[v + 1] + bias, 0.f));
        }
#else
        for (int nt = 0; nt < 8; ++nt) {
            const int n = nt * 16 + lm;
            const float w0 = We1[0 * 128 + n], w1 = We1[1 * 128 + n];
            const float w2 = We1[2 * 128 + n], w3 = We1[3 * 128 + n];
            const float bias = be1[n];
#pragma unroll
            for (int v = 0; v < 8; ++v) {
                const int row = v + 8 * half;
                const float* g = geo + row * 8;
                const float h0 = fmaxf(g[0] * w0 + g[1] * w1 + g[2] * w2 + g[3] * w3 + bias, 0.f);
                bufB[row * STRIDE + n] = f2bf(h0);
            }
        }
#endif
    }
    __syncthreads();

    // ---- Phase 4: edge MLP layer 2: e_emb = relu(h @ We2 + be2) -> bufA[:,128:256] ----
    if (active) {
        v16bf ah[4];                                  // whole 16x128 A-panel in registers
#pragma unroll
        for (int kt = 0; kt < 4; ++kt) ah[kt] = load_a_frag(bufB, lm, half, kt);
        for (int nt = 0; nt < 8; ++nt) {
            if (nt + 1 < 8) __builtin_prefetch(fWe2 + (nt + 1) * 512 + lane * 16, 0, 1);
            v8f acc = {};
#pragma unroll
            for (int kt = 0; kt < 4; ++kt) {
                v16bf b = load_b_frag(fWe2, kt * 8 + nt, lane);
                acc = __builtin_amdgcn_wmma_f32_16x16x32_bf16(false, ah[kt], false, b, (short)0, acc, false, false);
            }
            const int n = nt * 16 + lm;
            const float bias = be2[n];
#pragma unroll
            for (int v = 0; v < 8; v += 2)
                store_pair(bufA + (v + 0 + 8 * half) * STRIDE + 128 + n,
                           bufA + (v + 1 + 8 * half) * STRIDE + 128 + n,
                           fmaxf(acc[v] + bias, 0.f), fmaxf(acc[v + 1] + bias, 0.f));
        }
    }
    __syncthreads();

    // ---- Phase 5: attention logits + softmax (cooperative, vectorized) ----
    if (active) {
        const float* xi = x + (size_t)i * 128;
        // q . Wa[0:128] : one float4 chunk per lane + full-wave reduction
        const float4 xq = *(const float4*)(xi + lane * 4);
        const float4 wq = *(const float4*)(Wa + lane * 4);
        float qd = xq.x * wq.x + xq.y * wq.y + xq.z * wq.z + xq.w * wq.w;
#pragma unroll
        for (int off = 16; off; off >>= 1) qd += __shfl_xor(qd, off);
        // row dot: lane lm+16*half handles cols [half*128, half*128+128) of row lm
        const unsigned short* r = bufA + lm * STRIDE + half * 128;
        const float* wa = Wa + 128 + half * 128;
        float acc = 0.f;
#pragma unroll 4
        for (int k = 0; k < 128; k += 8) {
            const uint4 u = *(const uint4*)(r + k);           // 8 bf16
            const float4 w0 = *(const float4*)(wa + k);
            const float4 w1 = *(const float4*)(wa + k + 4);
            acc += bf2f_lo(u.x) * w0.x + bf2f_hi(u.x) * w0.y;
            acc += bf2f_lo(u.y) * w0.z + bf2f_hi(u.y) * w0.w;
            acc += bf2f_lo(u.z) * w1.x + bf2f_hi(u.z) * w1.y;
            acc += bf2f_lo(u.w) * w1.z + bf2f_hi(u.w) * w1.w;
        }
        acc += __shfl_xor(acc, 16);                           // combine the two half-row dots
        if (lane < 16) {
            float logit = qd + acc + ba[0];
            float mx = logit;
#pragma unroll
            for (int off = 8; off; off >>= 1) mx = fmaxf(mx, __shfl_xor(mx, off));
            const float ex = __expf(logit - mx);
            float sm = ex;
#pragma unroll
            for (int off = 8; off; off >>= 1) sm += __shfl_xor(sm, off);
            geo[lane * 8 + 4] = ex / sm;                      // alpha
        }
    }
    __syncthreads();

    // ---- Phase 6: GEMM1: t = relu(msg_in @ Wm1 + bm1)  (16x256 @ 256x256) -> bufB ----
    if (active) {
        v16bf am[8];                                  // whole 16x256 A-panel in registers (64 VGPRs)
#pragma unroll
        for (int kt = 0; kt < 8; ++kt) am[kt] = load_a_frag(bufA, lm, half, kt);
        for (int nt = 0; nt < 16; ++nt) {
            if (nt + 1 < 16) __builtin_prefetch(fWm1 + (nt + 1) * 512 + lane * 16, 0, 1);
            v8f acc = {};
#pragma unroll
            for (int kt = 0; kt < 8; ++kt) {
                v16bf b = load_b_frag(fWm1, kt * 16 + nt, lane);
                acc = __builtin_amdgcn_wmma_f32_16x16x32_bf16(false, am[kt], false, b, (short)0, acc, false, false);
            }
            const int n = nt * 16 + lm;
            const float bias = bm1[n];
#pragma unroll
            for (int v = 0; v < 8; v += 2)
                store_pair(bufB + (v + 0 + 8 * half) * STRIDE + n,
                           bufB + (v + 1 + 8 * half) * STRIDE + n,
                           fmaxf(acc[v] + bias, 0.f), fmaxf(acc[v + 1] + bias, 0.f));
        }
    }
    __syncthreads();

    // ---- Phase 7: GEMM2: msg = t @ Wm2 + bm2 ; agg = sum_m alpha[m]*msg[m] ; out = x + agg ----
    if (active) {
        float al[8];
#pragma unroll
        for (int v = 0; v < 8; ++v) al[v] = geo[(v + 8 * half) * 8 + 4];
        v16bf at[8];
#pragma unroll
        for (int kt = 0; kt < 8; ++kt) at[kt] = load_a_frag(bufB, lm, half, kt);
        const float* xi = x + (size_t)i * 128;
        for (int nt = 0; nt < 8; ++nt) {
            if (nt + 1 < 8) __builtin_prefetch(fWm2 + (nt + 1) * 512 + lane * 16, 0, 1);
            v8f acc = {};
#pragma unroll
            for (int kt = 0; kt < 8; ++kt) {
                v16bf b = load_b_frag(fWm2, kt * 8 + nt, lane);
                acc = __builtin_amdgcn_wmma_f32_16x16x32_bf16(false, at[kt], false, b, (short)0, acc, false, false);
            }
            const int n = nt * 16 + lm;
            const float bias = bm2[n];
            float s = 0.f;
#pragma unroll
            for (int v = 0; v < 8; ++v) s += al[v] * (acc[v] + bias);
            s += __shfl_xor(s, 16);                 // combine rows 0-7 with rows 8-15
            if (half == 0) out[(size_t)i * 128 + n] = xi[n] + s;
        }
    }
}

extern "C" void kernel_launch(void* const* d_in, const int* in_sizes, int n_in,
                              void* d_out, int out_size, void* d_ws, size_t ws_size,
                              hipStream_t stream) {
    const float* x     = (const float*)d_in[0];
    const int*   nbr   = (const int*)  d_in[1];
    const float* boxes = (const float*)d_in[2];
    const float* We1   = (const float*)d_in[3];
    const float* be1   = (const float*)d_in[4];
    const float* We2   = (const float*)d_in[5];
    const float* be2   = (const float*)d_in[6];
    const float* Wm1   = (const float*)d_in[7];
    const float* bm1   = (const float*)d_in[8];
    const float* Wm2   = (const float*)d_in[9];
    const float* bm2   = (const float*)d_in[10];
    const float* Wa    = (const float*)d_in[11];
    const float* ba    = (const float*)d_in[12];
    float* out = (float*)d_out;

    const int n = in_sizes[0] / 128;   // N nodes

    unsigned short* ws   = (unsigned short*)d_ws;
    unsigned short* fWe2 = ws;                 // (128/32)*(128/16)=32 tiles * 512
    unsigned short* fWm1 = ws + 16384;         // (256/32)*(256/16)=128 tiles * 512
    unsigned short* fWm2 = ws + 16384 + 65536; // (256/32)*(128/16)=64 tiles * 512

    swizzle_bf16_frag<<<(1024 + 255) / 256, 256, 0, stream>>>(We2, fWe2, 128, 128, 1024);
    swizzle_bf16_frag<<<(4096 + 255) / 256, 256, 0, stream>>>(Wm1, fWm1, 256, 256, 4096);
    swizzle_bf16_frag<<<(2048 + 255) / 256, 256, 0, stream>>>(Wm2, fWm2, 256, 128, 2048);

    const int smem = WAVES_PER_WG * WAVE_LDS_BYTES;   // 69632 B
    (void)hipFuncSetAttribute((const void*)fused_graph_msg,
                              hipFuncAttributeMaxDynamicSharedMemorySize, smem);
    fused_graph_msg<<<(n + WAVES_PER_WG - 1) / WAVES_PER_WG, 32 * WAVES_PER_WG, smem, stream>>>(
        x, nbr, boxes, We1, be1, be2, bm1, bm2, Wa, ba, fWe2, fWm1, fWm2, out, n);
}
